// _PointnetAdaptSAModuleBase_38774964748860
// MI455X (gfx1250) — compile-verified, matrix operand order checked by reading, not run on hardware
//
#include <hip/hip_runtime.h>
#include <hip/hip_bf16.h>

typedef __attribute__((ext_vector_type(16))) _Float16 v16h;
typedef __attribute__((ext_vector_type(8)))  _Float16 v8h;
typedef __attribute__((ext_vector_type(8)))  float    v8f;

constexpr int   B_ = 4, N_ = 16384, C_ = 32;
constexpr int   NPOINT = 1024, NSAMPLE = 32;
constexpr float R2 = 0.8f * 0.8f;
constexpr int   NCOL = NPOINT * NSAMPLE;   // 32768 columns per batch

// ---------------------------------------------------------------- FPS
// One workgroup per batch. Running min-distance lives in registers
// (16 floats/thread, 1024 threads). 1023 serial argmax steps via LDS tree.
__global__ void fps_kernel(const float* __restrict__ xyz, int* __restrict__ fps_idx) {
    const int b   = blockIdx.x;
    const int tid = threadIdx.x;                 // 0..1023
    const float* xb = xyz + (size_t)b * N_ * 3;

    float ldist[16];
#pragma unroll
    for (int q = 0; q < 16; ++q) ldist[q] = 1e10f;

    __shared__ float s_val[1024];
    __shared__ int   s_idx[1024];
    __shared__ int   s_last;
    if (tid == 0) { fps_idx[b * NPOINT + 0] = 0; s_last = 0; }
    __syncthreads();

    for (int it = 1; it < NPOINT; ++it) {
        const int last = s_last;
        const float cx = xb[last * 3 + 0];
        const float cy = xb[last * 3 + 1];
        const float cz = xb[last * 3 + 2];
        float best = -1.0f; int bidx = 0;
#pragma unroll
        for (int q = 0; q < 16; ++q) {
            const int i = tid + q * 1024;
            const float dx = xb[i * 3 + 0] - cx;
            const float dy = xb[i * 3 + 1] - cy;
            const float dz = xb[i * 3 + 2] - cz;
            const float d  = dx * dx + dy * dy + dz * dz;
            const float nd = fminf(ldist[q], d);
            ldist[q] = nd;
            if (nd > best) { best = nd; bidx = i; }   // strict > keeps first max
        }
        s_val[tid] = best; s_idx[tid] = bidx;
        __syncthreads();
        for (int off = 512; off > 0; off >>= 1) {
            if (tid < off) {
                const float ov = s_val[tid + off]; const int oi = s_idx[tid + off];
                if (ov > s_val[tid] || (ov == s_val[tid] && oi < s_idx[tid])) {
                    s_val[tid] = ov; s_idx[tid] = oi;
                }
            }
            __syncthreads();
        }
        if (tid == 0) { fps_idx[b * NPOINT + it] = s_idx[0]; s_last = s_idx[0]; }
        __syncthreads();
    }
}

__global__ void gather_newxyz(const float* __restrict__ xyz,
                              const int* __restrict__ fps_idx,
                              float* __restrict__ new_xyz) {
    const int t = blockIdx.x * blockDim.x + threadIdx.x;   // b*NPOINT + j
    if (t >= B_ * NPOINT) return;
    const int b = t / NPOINT;
    const int i = fps_idx[t];
    const float* p = xyz + ((size_t)b * N_ + i) * 3;
    new_xyz[t * 3 + 0] = p[0];
    new_xyz[t * 3 + 1] = p[1];
    new_xyz[t * 3 + 2] = p[2];
}

// ---------------------------------------------------------------- ball query
// One wave (32 lanes) per query. Ordered first-32-hits via ballot + prefix
// popcount; matches jax top_k(-scores) semantics incl. first-hit padding.
__global__ void ball_query(const float* __restrict__ xyz,
                           const float* __restrict__ centers,
                           int* __restrict__ idx) {
    const int gw   = (blockIdx.x * blockDim.x + threadIdx.x) >> 5;
    const int lane = threadIdx.x & 31;
    if (gw >= B_ * NPOINT) return;
    const int b = gw / NPOINT;
    const float* xb = xyz + (size_t)b * N_ * 3;
    const float cx = centers[gw * 3 + 0];
    const float cy = centers[gw * 3 + 1];
    const float cz = centers[gw * 3 + 2];
    int* out = idx + (size_t)gw * NSAMPLE;

    int cnt = 0, first = 0; bool any = false;
    for (int base = 0; base < N_ && cnt < NSAMPLE; base += 32) {
        const int i = base + lane;
        const float dx = xb[i * 3 + 0] - cx;
        const float dy = xb[i * 3 + 1] - cy;
        const float dz = xb[i * 3 + 2] - cz;
        const bool hit = (dx * dx + dy * dy + dz * dz) < R2;
        const unsigned mask = (unsigned)(__ballot(hit) & 0xffffffffull);
        if (mask) {
            if (!any) { any = true; first = base + __builtin_ctz(mask); }
            const unsigned below = mask & ((1u << lane) - 1u);
            const int slot = cnt + __popc(below);
            if (hit && slot < NSAMPLE) out[slot] = i;
            cnt += __popc(mask);
        }
    }
    const int total = cnt < NSAMPLE ? cnt : NSAMPLE;
    const int fill  = any ? first : 0;
    if (lane >= total) out[lane] = fill;
}

// ---------------------------------------------------------------- grouping
// Build channel-last f16 activation matrix G[b*NCOL + col][64]:
// ch 0..2 = xyz - center, ch 3..34 = features, ch 35..63 = 0 (K pad).
// Row is 128B-aligned -> written as 8x16B vector stores.
__global__ void group_kernel(const float* __restrict__ xyz,
                             const float* __restrict__ feat,
                             const float* __restrict__ centers,
                             const int* __restrict__ idx,
                             _Float16* __restrict__ G) {
    const int t = blockIdx.x * blockDim.x + threadIdx.x;   // b*NCOL + col
    if (t >= B_ * NCOL) return;
    const int b   = t / NCOL;
    const int col = t - b * NCOL;
    const int p   = col >> 5;
    const int i   = idx[t];
    const float* pt  = xyz + ((size_t)b * N_ + i) * 3;
    const float* cen = centers + ((size_t)b * NPOINT + p) * 3;
    const float* f   = feat + ((size_t)b * N_ + i) * C_;

    v8h row[8];
#pragma unroll
    for (int j = 0; j < 8; ++j)
#pragma unroll
        for (int v = 0; v < 8; ++v) row[j][v] = (_Float16)0.0f;

    row[0][0] = (_Float16)(pt[0] - cen[0]);
    row[0][1] = (_Float16)(pt[1] - cen[1]);
    row[0][2] = (_Float16)(pt[2] - cen[2]);
#pragma unroll
    for (int c = 0; c < C_; ++c) {
        const int ch = 3 + c;
        row[ch >> 3][ch & 7] = (_Float16)f[c];
    }
    _Float16* g = G + (size_t)t * 64;
#pragma unroll
    for (int j = 0; j < 8; ++j) *(v8h*)(g + 8 * j) = row[j];
}

// ---------------------------------------------------------------- weight packing
// One-shot: repack W[Cout][Cin] (f32) into the exact WMMA 16-bit A-operand
// lane/K swizzle, zero-padded: A[((ot*nCh + ch)*32 + lane)*16 + h].
__global__ void pack_weights(const float* __restrict__ W, _Float16* __restrict__ A,
                             int Cin, int Cout, int OTiles, int nCh) {
    const int t = blockIdx.x * blockDim.x + threadIdx.x;
    const int total = OTiles * nCh * 32 * 16;
    if (t >= total) return;
    const int h    = t & 15;
    const int lane = (t >> 4) & 31;
    const int cc   = t >> 9;            // ot*nCh + ch
    const int ch   = cc % nCh;
    const int ot   = cc / nCh;
    const int hi   = (lane >= 16) ? 1 : 0;
    const int k = ch * 32 + ((h < 8) ? h : (h + 8)) + (hi ? 8 : 0);
    const int m = ot * 16 + (lane & 15);
    const float wv = (k < Cin && m < Cout) ? W[m * Cin + k] : 0.0f;
    A[t] = (_Float16)wv;
}

// ---------------------------------------------------------------- WMMA GEMM
// One wave per (b, o_tile, p); covers both 16-column tiles of p's 32 samples.
// A operand: one 32B contiguous load per K-chunk (pre-swizzled).
// B operand: two 16B contiguous loads per K-chunk per tile (channel-last f16).
enum { EP_F16RELU = 0, EP_F16RELU_PAD = 1, EP_ST = 2, EP_MAXPOOL = 3 };

template <int MODE, int NCH, int COUT, int COUTPAD, int OTILES>
__global__ void gemm_wmma(const _Float16* __restrict__ X,      // [B*NCOL][NCH*32]
                          const _Float16* __restrict__ Apack,  // packed weights
                          const float* __restrict__ bias,      // [COUT]
                          _Float16* __restrict__ Y,            // f16 out (MODE 0/1)
                          float* __restrict__ Yf) {            // f32 out (MODE 2/3)
    constexpr int CINPAD = NCH * 32;
    const int gw   = (blockIdx.x * blockDim.x + threadIdx.x) >> 5;
    const int lane = threadIdx.x & 31;
    if (gw >= B_ * OTILES * NPOINT) return;
    const int b   = gw / (OTILES * NPOINT);
    const int rem = gw - b * (OTILES * NPOINT);
    const int ot  = rem / NPOINT;
    const int p   = rem - ot * NPOINT;
    const int obase = ot * 16;
    const int hi = (lane >= 16) ? 1 : 0;
    const int q  = lane & 15;
    const int rowbase = obase + (hi ? 8 : 0);

    v8f acc0, acc1;
#pragma unroll
    for (int v = 0; v < 8; ++v) {
        float bv;
        if constexpr (COUT % 16 == 0) {
            bv = bias[rowbase + v];
        } else {
            const int r = rowbase + v;
            bv = (r < COUT) ? bias[r] : 0.0f;
        }
        acc0[v] = bv; acc1[v] = bv;
    }

    const size_t colbase = (size_t)b * NCOL + (size_t)p * NSAMPLE;
    const _Float16* x0 = X + (colbase + q) * CINPAD + (hi ? 8 : 0);
    const _Float16* x1 = X + (colbase + 16 + q) * CINPAD + (hi ? 8 : 0);
    const _Float16* ap = Apack + ((size_t)ot * NCH * 32 + lane) * 16;

#pragma unroll
    for (int ch = 0; ch < NCH; ++ch) {
        const v16h a = *(const v16h*)(ap + (size_t)ch * 32 * 16);
        const int kbase = ch << 5;
        const v8h b0lo = *(const v8h*)(x0 + kbase);
        const v8h b0hi = *(const v8h*)(x0 + kbase + 16);
        const v8h b1lo = *(const v8h*)(x1 + kbase);
        const v8h b1hi = *(const v8h*)(x1 + kbase + 16);
        v16h bm0, bm1;
#pragma unroll
        for (int h = 0; h < 8; ++h) {
            bm0[h] = b0lo[h]; bm0[h + 8] = b0hi[h];
            bm1[h] = b1lo[h]; bm1[h + 8] = b1hi[h];
        }
        acc0 = __builtin_amdgcn_wmma_f32_16x16x32_f16(false, a, false, bm0,
                                                      (short)0, acc0, false, false);
        acc1 = __builtin_amdgcn_wmma_f32_16x16x32_f16(false, a, false, bm1,
                                                      (short)0, acc1, false, false);
    }

    if constexpr (MODE == EP_F16RELU || MODE == EP_F16RELU_PAD) {
        v8h y0, y1;
#pragma unroll
        for (int v = 0; v < 8; ++v) {
            y0[v] = (_Float16)fmaxf(acc0[v], 0.0f);
            y1[v] = (_Float16)fmaxf(acc1[v], 0.0f);
        }
        _Float16* yy0 = Y + (colbase + q) * COUTPAD + rowbase;
        _Float16* yy1 = Y + (colbase + 16 + q) * COUTPAD + rowbase;
        *(v8h*)yy0 = y0;
        *(v8h*)yy1 = y1;
        if constexpr (MODE == EP_F16RELU_PAD) {  // zero channels 16..31 (K pad)
            v8h z;
#pragma unroll
            for (int v = 0; v < 8; ++v) z[v] = (_Float16)0.0f;
            *(v8h*)(yy0 + 16) = z;
            *(v8h*)(yy1 + 16) = z;
        }
    } else if constexpr (MODE == EP_ST) {        // semantic_trans rows 0..2, f32
        if (!hi) {
#pragma unroll
            for (int v = 0; v < 3; ++v) {
                Yf[(colbase + q) * 3 + v]      = acc0[v];
                Yf[(colbase + 16 + q) * 3 + v] = acc1[v];
            }
        }
    } else {                                     // EP_MAXPOOL: relu + max over s
#pragma unroll
        for (int v = 0; v < 8; ++v) {
            float m = fmaxf(fmaxf(acc0[v], 0.0f), fmaxf(acc1[v], 0.0f));
            m = fmaxf(m, __shfl_xor(m, 1, 32));
            m = fmaxf(m, __shfl_xor(m, 2, 32));
            m = fmaxf(m, __shfl_xor(m, 4, 32));
            m = fmaxf(m, __shfl_xor(m, 8, 32));
            if (q == 0) {                        // lanes 0 (rows v) / 16 (rows v+8)
                const int r = rowbase + v;
                Yf[((size_t)b * COUT + r) * NPOINT + p] = m;
            }
        }
    }
}

// ---------------------------------------------------------------- node offset
__global__ void modxyz_kernel(const float* __restrict__ ST,
                              const _Float16* __restrict__ G,
                              const float* __restrict__ new_xyz,
                              float* __restrict__ mod_xyz,
                              float* __restrict__ out0) {
    const int t = blockIdx.x * blockDim.x + threadIdx.x;   // bp*3 + axis
    if (t >= B_ * NPOINT * 3) return;
    const int axis = t % 3;
    const int bp   = t / 3;
    const int b    = bp / NPOINT;
    const int p    = bp % NPOINT;
    const size_t colbase = (size_t)b * NCOL + (size_t)p * NSAMPLE;
    float s = 0.0f;
#pragma unroll
    for (int ss = 0; ss < NSAMPLE; ++ss) {
        const size_t col = colbase + ss;
        s += ST[col * 3 + axis] * (float)G[col * 64 + axis];
    }
    const float m = new_xyz[t] + s * (1.0f / NSAMPLE);
    mod_xyz[t] = m;
    out0[t] = fminf(fmaxf(m, -2.0f), 2.0f);
}

// ---------------------------------------------------------------- launch
extern "C" void kernel_launch(void* const* d_in, const int* in_sizes, int n_in,
                              void* d_out, int out_size, void* d_ws, size_t ws_size,
                              hipStream_t stream) {
    const float* xyz    = (const float*)d_in[0];
    const float* feat   = (const float*)d_in[1];
    const float* w_off0 = (const float*)d_in[2];
    const float* b_off0 = (const float*)d_in[3];
    const float* w_off1 = (const float*)d_in[4];
    const float* b_off1 = (const float*)d_in[5];
    const float* w0     = (const float*)d_in[6];
    const float* b0     = (const float*)d_in[7];
    const float* w1     = (const float*)d_in[8];
    const float* b1     = (const float*)d_in[9];
    const float* w2     = (const float*)d_in[10];
    const float* b2     = (const float*)d_in[11];

    char* ws = (char*)d_ws;
    int*      fpsIdx = (int*)(ws + 0);                      //  16 KB
    float*    newXyz = (float*)(ws + 16384);                //  48 KB
    float*    modXyz = (float*)(ws + 65536);                //  48 KB
    int*      idx    = (int*)(ws + 131072);                 // 512 KB
    float*    STbuf  = (float*)(ws + 655360);               // 1.5 MB
    _Float16* aOff0  = (_Float16*)(ws + 2228224);           //   2 KB (1ot x 2ch)
    _Float16* aOff1  = (_Float16*)(ws + 2230272);           //   1 KB (1ot x 1ch)
    _Float16* aW0    = (_Float16*)(ws + 2231296);           //   8 KB (4ot x 2ch)
    _Float16* aW1    = (_Float16*)(ws + 2239488);           //   8 KB (4ot x 2ch)
    _Float16* aW2    = (_Float16*)(ws + 2247680);           //  16 KB (8ot x 2ch)
    _Float16* G      = (_Float16*)(ws + (4ull << 20));      //  16 MB  [B*NCOL][64]
    _Float16* H0     = (_Float16*)(ws + 20971520ull);       //  16 MB  (also Hoff)
    _Float16* Hoff   = H0;                                  //  [B*NCOL][32], dead before H0 use
    _Float16* H1     = G;                                   //  alias: G dead after layer w0

    float* out_xyz  = (float*)d_out;           // (4,1024,3)
    float* out_feat = (float*)d_out + (size_t)B_ * NPOINT * 3;   // (4,128,1024)

    // 0. pre-swizzle all weight matrices into WMMA A-operand layout
    pack_weights<<<(1 * 2 * 512 + 255) / 256, 256, 0, stream>>>(w_off0, aOff0, 35, 16, 1, 2);
    pack_weights<<<(1 * 1 * 512 + 255) / 256, 256, 0, stream>>>(w_off1, aOff1, 16, 3, 1, 1);
    pack_weights<<<(4 * 2 * 512 + 255) / 256, 256, 0, stream>>>(w0, aW0, 35, 64, 4, 2);
    pack_weights<<<(4 * 2 * 512 + 255) / 256, 256, 0, stream>>>(w1, aW1, 64, 64, 4, 2);
    pack_weights<<<(8 * 2 * 512 + 255) / 256, 256, 0, stream>>>(w2, aW2, 64, 128, 8, 2);

    // 1. FPS + centers
    fps_kernel<<<B_, 1024, 0, stream>>>(xyz, fpsIdx);
    gather_newxyz<<<(B_ * NPOINT + 255) / 256, 256, 0, stream>>>(xyz, fpsIdx, newXyz);

    // 2. ball query A + group
    ball_query<<<(B_ * NPOINT * 32) / 256, 256, 0, stream>>>(xyz, newXyz, idx);
    group_kernel<<<(B_ * NCOL + 255) / 256, 256, 0, stream>>>(xyz, feat, newXyz, idx, G);

    // 3. offset head: 16ch relu conv -> 3ch conv -> node offset -> mod_xyz
    {
        const int wv = B_ * 1 * NPOINT;  // waves
        gemm_wmma<EP_F16RELU_PAD, 2, 16, 32, 1><<<(wv * 32) / 256, 256, 0, stream>>>(
            G, aOff0, b_off0, Hoff, nullptr);
        gemm_wmma<EP_ST, 1, 3, 0, 1><<<(wv * 32) / 256, 256, 0, stream>>>(
            Hoff, aOff1, b_off1, nullptr, STbuf);
    }
    modxyz_kernel<<<(B_ * NPOINT * 3 + 255) / 256, 256, 0, stream>>>(
        STbuf, G, newXyz, modXyz, out_xyz);

    // 4. ball query G (unclipped mod_xyz, per reference) + regroup
    ball_query<<<(B_ * NPOINT * 32) / 256, 256, 0, stream>>>(xyz, modXyz, idx);
    group_kernel<<<(B_ * NCOL + 255) / 256, 256, 0, stream>>>(xyz, feat, modXyz, idx, G);

    // 5. feature MLP: 64 -> 64 -> 128, fused maxpool in last epilogue
    {
        const int wv4 = B_ * 4 * NPOINT;
        gemm_wmma<EP_F16RELU, 2, 64, 64, 4><<<(wv4 * 32) / 256, 256, 0, stream>>>(
            G, aW0, b0, H0, nullptr);
        gemm_wmma<EP_F16RELU, 2, 64, 64, 4><<<(wv4 * 32) / 256, 256, 0, stream>>>(
            H0, aW1, b1, H1, nullptr);
        const int wv8 = B_ * 8 * NPOINT;
        gemm_wmma<EP_MAXPOOL, 2, 128, 0, 8><<<(wv8 * 32) / 256, 256, 0, stream>>>(
            H1, aW2, b2, nullptr, out_feat);
    }
}
